// mHCGPT_40518721470832
// MI455X (gfx1250) — matrix operand order repaired and checked by reading, not compiled, not run on hardware
//
#include <hip/hip_runtime.h>

// ---------------------------------------------------------------------------
// mHC-GPT forward for MI455X (gfx1250, wave32, WMMA).
// All GEMMs run as bf16 x bf16 -> f32 via v_wmma_f32_16x16x32_bf16.
// ---------------------------------------------------------------------------

using bf16 = __bf16;
typedef __attribute__((ext_vector_type(8)))  float v8f;
typedef __attribute__((ext_vector_type(16))) bf16  v16bf;
typedef __attribute__((ext_vector_type(8)))  bf16  v8bf;

#define EPI_F32    0
#define EPI_BF16   1
#define EPI_GELU   2
#define EPI_SCORES 3

// model dims
#define TOKS 2048      // B*S
#define DMODEL 768
#define NHEAD 12
#define HDIM 64
#define FFDIM 3072
#define VOCAB 50257

static __device__ __forceinline__ v8f wmma_bf16(v16bf a, v16bf b, v8f c) {
  return __builtin_amdgcn_wmma_f32_16x16x32_bf16(
      /*neg_a=*/false, a, /*neg_b=*/false, b,
      /*c_mod=*/(short)0, c, /*reuse_a=*/false, /*reuse_b=*/false);
}

// ---------------------------------------------------------------------------
// Generic WMMA GEMM:  C[m,n] = sum_k A[m,k] * BT[n,k]   (BT is (N,K) row-major)
// Block = 256 thr = 8 wave32; block tile 128(M) x 64(N); wave tile 32x32.
// Batched over z = b*nh + h with separate b/h strides for A, B, C.
// ---------------------------------------------------------------------------
__global__ __launch_bounds__(256) void k_gemm_bt(
    const bf16* __restrict__ A,  long long sAb, long long sAh, int lda,
    const bf16* __restrict__ Bt, long long sBb, long long sBh, int ldb,
    void* __restrict__ C,        long long sCb, long long sCh, int ldc,
    int M, int N, int K, int nh, int epi, float scale)
{
  const int z  = blockIdx.z;
  const int bb = z / nh;
  const int hh = z % nh;
  const bf16* Ab = A  + bb * sAb + hh * sAh;
  const bf16* Bb = Bt + bb * sBb + hh * sBh;

  const int mB = blockIdx.y * 128;
  const int nB = blockIdx.x * 64;
  const int wave = threadIdx.x >> 5;
  const int lane = threadIdx.x & 31;
  const int wm = wave >> 1;            // 0..3
  const int wn = wave & 1;             // 0..1
  const int mW = mB + wm * 32;
  const int nW = nB + wn * 32;
  const int rl  = lane & 15;
  const int q8  = (lane >> 4) * 8;     // unused (A loads use k16 as well)
  const int k16 = (lane >> 4) * 16;    // B: per-lane K half offset

  v8f c00 = {}, c01 = {}, c10 = {}, c11 = {};

  const bool skip = (epi == EPI_SCORES) && (nB > mB + 127); // fully causal-masked tile
  if (!skip) {
    const bf16* a0p = Ab + (long long)(mW + rl) * lda;
    const bf16* a1p = a0p + (long long)16 * lda;
    int r0 = nW + rl;      if (r0 > N - 1) r0 = N - 1;
    int r1 = nW + 16 + rl; if (r1 > N - 1) r1 = N - 1;
    const bf16* b0p = Bb + (long long)r0 * ldb + k16;
    const bf16* b1p = Bb + (long long)r1 * ldb + k16;

    for (int k = 0; k < K; k += 32) {
      // A fragment (contiguous K: lanes 0-15 K=0..15; lanes 16-31 K=16..31)
      v8bf a0l = *(const v8bf*)(a0p + k + k16);
      v8bf a0h = *(const v8bf*)(a0p + k + k16 + 8);
      v8bf a1l = *(const v8bf*)(a1p + k + k16);
      v8bf a1h = *(const v8bf*)(a1p + k + k16 + 8);
      v16bf fa0 = __builtin_shufflevector(a0l, a0h, 0,1,2,3,4,5,6,7,8,9,10,11,12,13,14,15);
      v16bf fa1 = __builtin_shufflevector(a1l, a1h, 0,1,2,3,4,5,6,7,8,9,10,11,12,13,14,15);
      // B fragment (lane = N column from transposed weight row, 16 contiguous K)
      v16bf fb0 = *(const v16bf*)(b0p + k);
      v16bf fb1 = *(const v16bf*)(b1p + k);
      __builtin_prefetch(b0p + k + 256, 0, 1);   // global_prefetch_b8 on weight stream

      c00 = wmma_bf16(fa0, fb0, c00);
      c01 = wmma_bf16(fa0, fb1, c01);
      c10 = wmma_bf16(fa1, fb0, c10);
      c11 = wmma_bf16(fa1, fb1, c11);
    }
  }

  // Epilogue. C/D layout: VGPR r -> M = r + 8*(lane>=16), N = lane&15.
  const int mh   = (lane >> 4) * 8;
  const int col0 = nW + rl;
  const int col1 = nW + 16 + rl;
  float* Cf = (float*)C + bb * sCb + hh * sCh;
  bf16*  Ch = (bf16*)C  + bb * sCb + hh * sCh;

#pragma unroll
  for (int r = 0; r < 8; r++) {
    const int row0 = mW + mh + r;
    const int row1 = row0 + 16;
    float vals[4] = { c00[r], c01[r], c10[r], c11[r] };
    int   rows[4] = { row0, row0, row1, row1 };
    int   cols[4] = { col0, col1, col0, col1 };
#pragma unroll
    for (int q = 0; q < 4; q++) {
      const int rr = rows[q], cc = cols[q];
      if (cc >= N) continue;
      const long long off = (long long)rr * ldc + cc;
      const float v = vals[q];
      if      (epi == EPI_F32)  Cf[off] = v;
      else if (epi == EPI_BF16) Ch[off] = (bf16)v;
      else if (epi == EPI_GELU) Ch[off] = (bf16)(0.5f * v * (1.0f + erff(v * 0.70710678f)));
      else                      Cf[off] = (cc > rr) ? -1e30f : v * scale;  // causal scores
    }
  }
}

// ---------------------------------------------------------------------------
// Embedding: x[b,s,n,d] = tok_emb[ids[b,s],d] + pos_emb[s,d], broadcast n=0..3
// ---------------------------------------------------------------------------
__global__ void k_embed(const int* __restrict__ ids, const float* __restrict__ tok,
                        const float* __restrict__ pos, float* __restrict__ x)
{
  int idx = blockIdx.x * 256 + threadIdx.x;        // token*768 + d
  if (idx >= TOKS * DMODEL) return;
  int t = idx / DMODEL, d = idx - t * DMODEL;
  int s = t & 1023;
  float v = tok[(long long)ids[t] * DMODEL + d] + pos[(long long)s * DMODEL + d];
  float* xr = x + (long long)t * (4 * DMODEL) + d;
  xr[0] = v; xr[768] = v; xr[1536] = v; xr[2304] = v;
}

// ---------------------------------------------------------------------------
// mhc "pre" stage, one block (128 thr) per token:
//   RMS-norm of flat x (3072), logits = xn@W*alpha + b (3072x24 matvec),
//   Sinkhorn(10) on 4x4 -> Hm, softmax -> pre, sigmoid -> post,
//   h[d] = sum_n pre[n]*x[n,d].  coef[t] = {Hm[16], pre[4], post[4]}.
// ---------------------------------------------------------------------------
__global__ __launch_bounds__(128) void k_mhc_pre(
    const float* __restrict__ x, const float* __restrict__ W,
    const float* __restrict__ bvec, const float* __restrict__ alphap,
    float* __restrict__ coef, float* __restrict__ h)
{
  __shared__ float xs[3072];
  __shared__ float part[128 * 24];
  __shared__ float red[128];
  __shared__ float cf[24];
  const int t = blockIdx.x, tid = threadIdx.x;
  const float* xrow = x + (long long)t * 3072;

  float ss = 0.f;
  for (int i = tid; i < 3072; i += 128) { float v = xrow[i]; xs[i] = v; ss += v * v; }
  red[tid] = ss; __syncthreads();
  for (int s = 64; s > 0; s >>= 1) { if (tid < s) red[tid] += red[tid + s]; __syncthreads(); }
  const float rms = rsqrtf(red[0] * (1.0f / 3072.0f) + 1e-6f);

  float acc[24];
#pragma unroll
  for (int j = 0; j < 24; j++) acc[j] = 0.f;
  for (int i = tid; i < 3072; i += 128) {
    const float v = xs[i] * rms;
    const float* wr = W + (long long)i * 24;
#pragma unroll
    for (int j = 0; j < 24; j++) acc[j] += v * wr[j];
  }
#pragma unroll
  for (int j = 0; j < 24; j++) part[tid * 24 + j] = acc[j];
  __syncthreads();

  const float alpha = alphap[0];
  if (tid < 24) {
    float s = 0.f;
    for (int k = 0; k < 128; k++) s += part[k * 24 + tid];
    cf[tid] = s * alpha + bvec[tid];                 // logits
  }
  __syncthreads();

  if (tid == 0) {
    float lm[16];
#pragma unroll
    for (int i = 0; i < 16; i++) lm[i] = cf[i];
    for (int it = 0; it < 10; it++) {
      for (int r = 0; r < 4; r++) {                  // rows (axis=-1)
        float m = lm[r*4];
        for (int c2 = 1; c2 < 4; c2++) m = fmaxf(m, lm[r*4 + c2]);
        float s = 0.f;
        for (int c2 = 0; c2 < 4; c2++) s += __expf(lm[r*4 + c2] - m);
        const float l = m + __logf(s);
        for (int c2 = 0; c2 < 4; c2++) lm[r*4 + c2] -= l;
      }
      for (int c2 = 0; c2 < 4; c2++) {               // cols (axis=-2)
        float m = lm[c2];
        for (int r = 1; r < 4; r++) m = fmaxf(m, lm[r*4 + c2]);
        float s = 0.f;
        for (int r = 0; r < 4; r++) s += __expf(lm[r*4 + c2] - m);
        const float l = m + __logf(s);
        for (int r = 0; r < 4; r++) lm[r*4 + c2] -= l;
      }
    }
    float pre[4], post[4];
    {
      float m = fmaxf(fmaxf(cf[16], cf[17]), fmaxf(cf[18], cf[19]));
      float s = 0.f;
      for (int n = 0; n < 4; n++) { pre[n] = __expf(cf[16 + n] - m); s += pre[n]; }
      const float inv = 1.f / s;
      for (int n = 0; n < 4; n++) pre[n] *= inv;
    }
    for (int n = 0; n < 4; n++) post[n] = 1.f / (1.f + __expf(-cf[20 + n]));
    float* cg = coef + (long long)t * 24;
    for (int i = 0; i < 16; i++) { const float hm = __expf(lm[i]); cf[i] = hm; cg[i] = hm; }
    for (int n = 0; n < 4; n++) { cf[16+n] = pre[n]; cg[16+n] = pre[n];
                                  cf[20+n] = post[n]; cg[20+n] = post[n]; }
  }
  __syncthreads();

  const float p0 = cf[16], p1 = cf[17], p2 = cf[18], p3 = cf[19];
  float* hrow = h + (long long)t * DMODEL;
  for (int d = tid; d < DMODEL; d += 128)
    hrow[d] = p0 * xs[d] + p1 * xs[768 + d] + p2 * xs[1536 + d] + p3 * xs[2304 + d];
}

// ---------------------------------------------------------------------------
// LayerNorm over D=768, output bf16 (GEMM-ready)
// ---------------------------------------------------------------------------
__global__ __launch_bounds__(256) void k_layernorm(
    const float* __restrict__ h, const float* __restrict__ g,
    const float* __restrict__ bb, bf16* __restrict__ out)
{
  __shared__ float red[256], red2[256];
  const int t = blockIdx.x, tid = threadIdx.x;
  const float* row = h + (long long)t * DMODEL;
  float s = 0.f, s2 = 0.f;
  for (int d = tid; d < DMODEL; d += 256) { const float v = row[d]; s += v; s2 += v * v; }
  red[tid] = s; red2[tid] = s2; __syncthreads();
  for (int st = 128; st > 0; st >>= 1) {
    if (tid < st) { red[tid] += red[tid + st]; red2[tid] += red2[tid + st]; }
    __syncthreads();
  }
  const float mu  = red[0] * (1.0f / DMODEL);
  const float var = red2[0] * (1.0f / DMODEL) - mu * mu;
  const float inv = rsqrtf(var + 1e-5f);
  bf16* orow = out + (long long)t * DMODEL;
  for (int d = tid; d < DMODEL; d += 256)
    orow[d] = (bf16)((row[d] - mu) * inv * g[d] + bb[d]);
}

// contract over n (sum 4 copies) + final LayerNorm -> bf16
__global__ __launch_bounds__(256) void k_contract_lnf(
    const float* __restrict__ x, const float* __restrict__ g,
    const float* __restrict__ bb, bf16* __restrict__ out)
{
  __shared__ float buf[DMODEL];
  __shared__ float red[256], red2[256];
  const int t = blockIdx.x, tid = threadIdx.x;
  const float* xr = x + (long long)t * 3072;
  float s = 0.f, s2 = 0.f;
  for (int d = tid; d < DMODEL; d += 256) {
    const float v = xr[d] + xr[768 + d] + xr[1536 + d] + xr[2304 + d];
    buf[d] = v; s += v; s2 += v * v;
  }
  red[tid] = s; red2[tid] = s2; __syncthreads();
  for (int st = 128; st > 0; st >>= 1) {
    if (tid < st) { red[tid] += red[tid + st]; red2[tid] += red2[tid + st]; }
    __syncthreads();
  }
  const float mu  = red[0] * (1.0f / DMODEL);
  const float var = red2[0] * (1.0f / DMODEL) - mu * mu;
  const float inv = rsqrtf(var + 1e-5f);
  bf16* orow = out + (long long)t * DMODEL;
  for (int d = tid; d < DMODEL; d += 256)
    orow[d] = (bf16)((buf[d] - mu) * inv * g[d] + bb[d]);
}

// x[n,d] <- sum_m Hm[n,m]*x[m,d] + post[n]*y[d]   (in-place safe per thread)
__global__ __launch_bounds__(256) void k_mhc_combine(
    float* __restrict__ x, const float* __restrict__ coef, const float* __restrict__ y)
{
  __shared__ float cf[24];
  const int t = blockIdx.x, tid = threadIdx.x;
  if (tid < 24) cf[tid] = coef[(long long)t * 24 + tid];
  __syncthreads();
  float* xr = x + (long long)t * 3072;
  const float* yr = y + (long long)t * DMODEL;
  for (int d = tid; d < DMODEL; d += 256) {
    const float x0 = xr[d], x1 = xr[768 + d], x2 = xr[1536 + d], x3 = xr[2304 + d];
    const float yv = yr[d];
    xr[d]        = cf[0] *x0 + cf[1] *x1 + cf[2] *x2 + cf[3] *x3 + cf[20] * yv;
    xr[768 + d]  = cf[4] *x0 + cf[5] *x1 + cf[6] *x2 + cf[7] *x3 + cf[21] * yv;
    xr[1536 + d] = cf[8] *x0 + cf[9] *x1 + cf[10]*x2 + cf[11]*x3 + cf[22] * yv;
    xr[2304 + d] = cf[12]*x0 + cf[13]*x1 + cf[14]*x2 + cf[15]*x3 + cf[23] * yv;
  }
}

// row softmax over 1024 (causal rows already masked to -1e30) -> bf16 P
__global__ __launch_bounds__(256) void k_softmax(const float* __restrict__ S,
                                                 bf16* __restrict__ P)
{
  __shared__ float red[256];
  const long long base = (long long)blockIdx.x * 1024;
  const int tid = threadIdx.x;
  float v[4];
#pragma unroll
  for (int i = 0; i < 4; i++) v[i] = S[base + tid + i * 256];
  float m = fmaxf(fmaxf(v[0], v[1]), fmaxf(v[2], v[3]));
  red[tid] = m; __syncthreads();
  for (int st = 128; st > 0; st >>= 1) { if (tid < st) red[tid] = fmaxf(red[tid], red[tid + st]); __syncthreads(); }
  m = red[0]; __syncthreads();
  float s = 0.f;
#pragma unroll
  for (int i = 0; i < 4; i++) { v[i] = __expf(v[i] - m); s += v[i]; }
  red[tid] = s; __syncthreads();
  for (int st = 128; st > 0; st >>= 1) { if (tid < st) red[tid] += red[tid + st]; __syncthreads(); }
  const float inv = 1.f / red[0];
#pragma unroll
  for (int i = 0; i < 4; i++) P[base + tid + i * 256] = (bf16)(v[i] * inv);
}

// f32 (K,N) -> bf16 (N,K)  (weight transpose+convert, LDS tiled)
__global__ void k_tcvt(const float* __restrict__ in, bf16* __restrict__ out, int K, int N)
{
  __shared__ float tile[32][33];
  const int n0 = blockIdx.x * 32, k0 = blockIdx.y * 32;
  const int tx = threadIdx.x, ty = threadIdx.y;
  for (int i = ty; i < 32; i += 8)
    tile[i][tx] = in[(long long)(k0 + i) * N + n0 + tx];
  __syncthreads();
  for (int i = ty; i < 32; i += 8)
    out[(long long)(n0 + i) * K + k0 + tx] = (bf16)tile[tx][i];
}

// f32 -> bf16 elementwise (tok_emb for the tied lm head)
__global__ void k_cvt(const float* __restrict__ in, bf16* __restrict__ out, long long n)
{
  const long long i = (long long)blockIdx.x * 256 + threadIdx.x;
  if (i < n) out[i] = (bf16)in[i];
}

// V slice of qkv (bf16) -> V^T (B,NH,HD,S) bf16 for the PV GEMM B-operand
__global__ void k_vT(const bf16* __restrict__ qkv, bf16* __restrict__ vT)
{
  const int idx = blockIdx.x * 256 + threadIdx.x;      // ((z*64+d)*1024+s)
  if (idx >= 2 * NHEAD * HDIM * 1024) return;
  const int s = idx & 1023;
  const int rest = idx >> 10;
  const int d = rest & 63;
  const int z = rest >> 6;
  const int b = z / NHEAD, hh = z % NHEAD;
  vT[idx] = qkv[((long long)(b * 1024 + s)) * 2304 + 1536 + hh * 64 + d];
}

// ---------------------------------------------------------------------------
// Workspace layout (bytes) — peak ~230 MB
// ---------------------------------------------------------------------------
static constexpr size_t OFF_X    = 0;                         // f32 (2048,4,768)  25.2MB
static constexpr size_t OFF_H    = OFF_X    + 25165824;       // f32 (2048,768)
static constexpr size_t OFF_HN   = OFF_H    + 6291456;        // bf16 (2048,768)
static constexpr size_t OFF_Y    = OFF_HN   + 3145728;        // f32 (2048,768)
static constexpr size_t OFF_COEF = OFF_Y    + 6291456;        // f32 (2048,24)
static constexpr size_t OFF_QKV  = OFF_COEF + 196608;         // bf16 (2048,2304)
static constexpr size_t OFF_VT   = OFF_QKV  + 9437184;        // bf16 (24,64,1024)
static constexpr size_t OFF_OBF  = OFF_VT   + 3145728;        // bf16 (2048,768)
static constexpr size_t OFF_A1   = OFF_OBF  + 3145728;        // bf16 (2048,3072)
static constexpr size_t OFF_W1T  = OFF_A1   + 12582912;       // bf16 weight slot 1
static constexpr size_t OFF_W2T  = OFF_W1T  + 4718592;        // bf16 weight slot 2
static constexpr size_t OFF_SC   = OFF_W2T  + 4718592;        // f32 scores (24,1024,1024) 100.7MB
static constexpr size_t OFF_P    = OFF_SC   + 100663296;      // bf16 probs 50.3MB
static constexpr size_t OFF_TOKB = OFF_SC;                    // bf16 tok_emb (reuses scores)

static inline void launch_gemm(hipStream_t st,
    const bf16* A, long long sAb, long long sAh, int lda,
    const bf16* Bt, long long sBb, long long sBh, int ldb,
    void* C, long long sCb, long long sCh, int ldc,
    int M, int N, int K, int batches, int nh, int epi, float scale)
{
  dim3 grid((N + 63) / 64, M / 128, batches);
  k_gemm_bt<<<grid, 256, 0, st>>>(A, sAb, sAh, lda, Bt, sBb, sBh, ldb,
                                  C, sCb, sCh, ldc, M, N, K, nh, epi, scale);
}

extern "C" void kernel_launch(void* const* d_in, const int* in_sizes, int n_in,
                              void* d_out, int out_size, void* d_ws, size_t ws_size,
                              hipStream_t stream)
{
  const int*   ids   = (const int*)  d_in[0];
  const float* tok   = (const float*)d_in[1];
  const float* pos   = (const float*)d_in[2];
  const float* ln1g  = (const float*)d_in[3];
  const float* ln1b  = (const float*)d_in[4];
  const float* ln2g  = (const float*)d_in[5];
  const float* ln2b  = (const float*)d_in[6];
  const float* Wqkv  = (const float*)d_in[7];
  const float* Wproj = (const float*)d_in[8];
  const float* Wfc1  = (const float*)d_in[9];
  const float* Wfc2  = (const float*)d_in[10];
  const float* mAW   = (const float*)d_in[11];
  const float* mAb   = (const float*)d_in[12];
  const float* mAal  = (const float*)d_in[13];
  const float* mFW   = (const float*)d_in[14];
  const float* mFb   = (const float*)d_in[15];
  const float* mFal  = (const float*)d_in[16];
  const float* lnfg  = (const float*)d_in[17];
  const float* lnfb  = (const float*)d_in[18];

  char* ws = (char*)d_ws;
  float* X    = (float*)(ws + OFF_X);
  float* H    = (float*)(ws + OFF_H);
  bf16*  HN   = (bf16*) (ws + OFF_HN);
  float* Y    = (float*)(ws + OFF_Y);
  float* COEF = (float*)(ws + OFF_COEF);
  bf16*  QKVb = (bf16*) (ws + OFF_QKV);
  bf16*  VT   = (bf16*) (ws + OFF_VT);
  bf16*  OBF  = (bf16*) (ws + OFF_OBF);
  bf16*  A1   = (bf16*) (ws + OFF_A1);
  bf16*  W1T  = (bf16*) (ws + OFF_W1T);
  bf16*  W2T  = (bf16*) (ws + OFF_W2T);
  float* SC   = (float*)(ws + OFF_SC);
  bf16*  P    = (bf16*) (ws + OFF_P);
  bf16*  TOKB = (bf16*) (ws + OFF_TOKB);

  const dim3 tb(32, 8);

  // embedding + expand to 4 hyper-streams
  k_embed<<<(TOKS * DMODEL) / 256, 256, 0, stream>>>(ids, tok, pos, X);

  for (int l = 0; l < 4; l++) {
    const float* Wqkv_l  = Wqkv  + (long long)l * DMODEL * 2304;
    const float* Wproj_l = Wproj + (long long)l * DMODEL * DMODEL;
    const float* Wfc1_l  = Wfc1  + (long long)l * DMODEL * FFDIM;
    const float* Wfc2_l  = Wfc2  + (long long)l * FFDIM * DMODEL;

    // ---------------- attention sub-block ----------------
    k_mhc_pre<<<TOKS, 128, 0, stream>>>(X, mAW + (long long)l * 3072 * 24,
                                        mAb + l * 24, mAal + l, COEF, H);
    k_layernorm<<<TOKS, 256, 0, stream>>>(H, ln1g + l * DMODEL, ln1b + l * DMODEL, HN);
    k_tcvt<<<dim3(2304 / 32, DMODEL / 32), tb, 0, stream>>>(Wqkv_l, W1T, DMODEL, 2304);
    launch_gemm(stream, HN, 0, 0, DMODEL, W1T, 0, 0, DMODEL,
                QKVb, 0, 0, 2304, TOKS, 2304, DMODEL, 1, 1, EPI_BF16, 0.f);
    k_vT<<<(2 * NHEAD * HDIM * 1024) / 256, 256, 0, stream>>>(QKVb, VT);
    // scores = Q K^T * 1/sqrt(64), causal masked
    launch_gemm(stream,
                QKVb,        (long long)1024 * 2304, 64, 2304,
                QKVb + 768,  (long long)1024 * 2304, 64, 2304,
                SC, (long long)NHEAD * 1024 * 1024, (long long)1024 * 1024, 1024,
                1024, 1024, 64, 2 * NHEAD, NHEAD, EPI_SCORES, 0.125f);
    k_softmax<<<2 * NHEAD * 1024, 256, 0, stream>>>(SC, P);
    // O = P V  (B operand = V^T rows)
    launch_gemm(stream,
                P,  (long long)NHEAD * 1024 * 1024, (long long)1024 * 1024, 1024,
                VT, (long long)NHEAD * HDIM * 1024, (long long)HDIM * 1024, 1024,
                OBF, (long long)1024 * DMODEL, 64, DMODEL,
                1024, HDIM, 1024, 2 * NHEAD, NHEAD, EPI_BF16, 0.f);
    k_tcvt<<<dim3(DMODEL / 32, DMODEL / 32), tb, 0, stream>>>(Wproj_l, W2T, DMODEL, DMODEL);
    launch_gemm(stream, OBF, 0, 0, DMODEL, W2T, 0, 0, DMODEL,
                Y, 0, 0, DMODEL, TOKS, DMODEL, DMODEL, 1, 1, EPI_F32, 0.f);
    k_mhc_combine<<<TOKS, 256, 0, stream>>>(X, COEF, Y);

    // ---------------- FFN sub-block ----------------
    k_mhc_pre<<<TOKS, 128, 0, stream>>>(X, mFW + (long long)l * 3072 * 24,
                                        mFb + l * 24, mFal + l, COEF, H);
    k_layernorm<<<TOKS, 256, 0, stream>>>(H, ln2g + l * DMODEL, ln2b + l * DMODEL, HN);
    k_tcvt<<<dim3(FFDIM / 32, DMODEL / 32), tb, 0, stream>>>(Wfc1_l, W1T, DMODEL, FFDIM);
    launch_gemm(stream, HN, 0, 0, DMODEL, W1T, 0, 0, DMODEL,
                A1, 0, 0, FFDIM, TOKS, FFDIM, DMODEL, 1, 1, EPI_GELU, 0.f);
    k_tcvt<<<dim3(DMODEL / 32, FFDIM / 32), tb, 0, stream>>>(Wfc2_l, W2T, FFDIM, DMODEL);
    launch_gemm(stream, A1, 0, 0, FFDIM, W2T, 0, 0, FFDIM,
                Y, 0, 0, DMODEL, TOKS, DMODEL, FFDIM, 1, 1, EPI_F32, 0.f);
    k_mhc_combine<<<TOKS, 256, 0, stream>>>(X, COEF, Y);
  }

  // contract + final LN, then tied lm-head (tok_emb is already (V,D) = B^T layout)
  k_contract_lnf<<<TOKS, 256, 0, stream>>>(X, lnfg, lnfb, HN);
  k_cvt<<<(int)(((long long)VOCAB * DMODEL + 255) / 256), 256, 0, stream>>>(
      tok, TOKB, (long long)VOCAB * DMODEL);
  launch_gemm(stream, HN, 0, 0, DMODEL, TOKB, 0, 0, DMODEL,
              d_out, 0, 0, VOCAB, TOKS, VOCAB, DMODEL, 1, 1, EPI_F32, 0.f);

  (void)in_sizes; (void)n_in; (void)out_size; (void)ws_size;
}
